// LearningModel_76484777607762
// MI455X (gfx1250) — matrix operand order repaired and checked by reading, not compiled, not run on hardware
//
#include <hip/hip_runtime.h>
#include <hip/hip_bf16.h>
#include <math.h>

// ---------------- problem constants (from reference) ----------------
#define KD    256          // embedding dim
#define KR    16           // rules
#define KT    500          // axiom types
#define KN0   8192         // init nodes
#define KL    6            // layers
#define KNL   8192         // nodes per layer
#define KB    (KNL / KR)   // 512 nodes per rule
#define KTOTAL (KN0 + KL * KNL)  // 57344

#define MB    32           // M rows per workgroup block (per rule)

typedef __attribute__((ext_vector_type(16))) _Float16 v16h;
typedef __attribute__((ext_vector_type(8)))  _Float16 v8h;
typedef __attribute__((ext_vector_type(8)))  float    v8f;

union V16U { v16h v; v8h h[2]; };

// ---------------- prep: weight convert + transpose to f16 ----------------
// W1 [R, 512, 256] f32  ->  W1hT [R, 256(out), 512(in)] f16
__global__ __launch_bounds__(256) void k_conv_w1(const float* __restrict__ W1,
                                                 _Float16* __restrict__ W1hT) {
  int idx = blockIdx.x * 256 + threadIdx.x;            // over R*256*512
  if (idx >= KR * 256 * 512) return;
  int i = idx & 511;              // input channel
  int o = (idx >> 9) & 255;       // output channel
  int r = idx >> 17;
  W1hT[idx] = (_Float16)W1[((size_t)r * 512 + i) * 256 + o];
}

// W2 [R, 256, 256] f32 -> W2hT [R, 256(out), 256(in)] f16
__global__ __launch_bounds__(256) void k_conv_w2(const float* __restrict__ W2,
                                                 _Float16* __restrict__ W2hT) {
  int idx = blockIdx.x * 256 + threadIdx.x;            // over R*256*256
  if (idx >= KR * 256 * 256) return;
  int i = idx & 255;
  int o = (idx >> 8) & 255;
  int r = idx >> 16;
  W2hT[idx] = (_Float16)W2[((size_t)r * 256 + i) * 256 + o];
}

// storeh[0:N0] = f16(init_table[thax_ids])
__global__ __launch_bounds__(256) void k_init_store(const int* __restrict__ thax,
                                                    const float* __restrict__ table,
                                                    _Float16* __restrict__ storeh) {
  int idx = blockIdx.x * 256 + threadIdx.x;            // over N0*256
  if (idx >= KN0 * KD) return;
  int c = idx & (KD - 1);
  int n = idx >> 8;
  storeh[idx] = (_Float16)table[(size_t)thax[n] * KD + c];
}

__global__ void k_zero_sums(float* sums) {
  int t = threadIdx.x;
  if (t < 5) sums[t] = 0.0f;
}

// ---------------- layer kernel: async gather + GEMM1(+ReLU) + GEMM2 ----------------
// grid = (R, KB/MB) = (16, 16); block = 256 (8 waves)
__global__ __launch_bounds__(256) void k_layer(
    const int* __restrict__ par,            // par_idx for this layer [NL,2]
    const _Float16* __restrict__ storeh,    // full f16 store (read)
    _Float16* __restrict__ outh,            // storeh + node_base*D (write)
    const _Float16* __restrict__ W1hT,      // [R,256,512]
    const float* __restrict__ b1,           // [R,256]
    const _Float16* __restrict__ W2hT,      // [R,256,256]
    const float* __restrict__ b2)           // [R,256]
{
  __shared__ __align__(16) _Float16 As[MB][512];   // gathered concat parents (32 KB)
  __shared__ __align__(16) _Float16 Hs[MB][256];   // hidden activations (16 KB)

  const int r    = blockIdx.x;          // rule
  const int mb   = blockIdx.y;          // m block within rule
  const int tid  = threadIdx.x;
  const int lane = tid & 31;
  const int wave = tid >> 5;
  const int ln   = lane & 15;           // N/M index within tile
  const int half = lane >> 4;           // lane-half selector
  const int rowBase = mb * MB;          // row offset within rule

  // ---- phase 0: async gather of parent embeddings directly into LDS ----
  // GLOBAL_LOAD_ASYNC_TO_LDS_B128: 16B per lane, global -> LDS, tracked by ASYNCcnt.
  {
    unsigned ldsBase = (unsigned)(uintptr_t)(&As[0][0]);   // low 32 bits = LDS offset
    for (int idx = tid; idx < MB * 64; idx += 256) {
      int row   = idx >> 6;             // 0..MB-1
      int chunk = idx & 63;             // 64 x 16B = 1024B = 512 halves
      int nodeLocal = r * KB + rowBase + row;
      int sel = chunk >> 5;             // parent 0 (first 256 halves) or 1
      int p   = par[nodeLocal * 2 + sel];
      unsigned long long gaddr =
          (unsigned long long)(uintptr_t)(storeh + (size_t)p * KD) +
          (unsigned long long)((chunk & 31) * 16);
      unsigned laddr = ldsBase + (unsigned)(row * 1024 + chunk * 16);
      asm volatile("global_load_async_to_lds_b128 %0, %1, off"
                   :: "v"(laddr), "v"(gaddr) : "memory");
    }
    asm volatile("s_wait_asynccnt 0x0" ::: "memory");
  }
  __syncthreads();

  // ---- phase 1: H = relu(A @ W1 + b1) ----
  // wave -> (mt, group of 4 nt); A fragment shared across the 4 WMMAs per K-step
  const _Float16* w1r = W1hT + (size_t)r * 256 * 512;
  const float*    b1r = b1 + r * KD;
  {
    const int mt  = wave >> 2;                 // 0..1
    const int nt0 = (wave & 3) * 4;            // 0,4,8,12
    v8f acc[4] = {{}, {}, {}, {}};
    for (int kb = 0; kb < 512; kb += 32) {
      V16U ua;   // A in contiguous k order (same as B): lo = K[kb+16h..+7], hi = K[kb+16h+8..+15]
      ua.h[0] = *(const v8h*)(&As[mt * 16 + ln][kb + 16 * half]);
      ua.h[1] = *(const v8h*)(&As[mt * 16 + ln][kb + 16 * half + 8]);
      const _Float16* bbase = w1r + (size_t)(nt0 * 16 + ln) * 512 + kb + 16 * half;
      #pragma unroll
      for (int j = 0; j < 4; j++) {
        v16h bfrag = *(const v16h*)(bbase + (size_t)j * 16 * 512);
        acc[j] = __builtin_amdgcn_wmma_f32_16x16x32_f16(
            false, ua.v, false, bfrag, (short)0, acc[j], false, false);
      }
    }
    #pragma unroll
    for (int j = 0; j < 4; j++) {
      int nt = nt0 + j;
      float bias = b1r[nt * 16 + ln];
      #pragma unroll
      for (int v = 0; v < 8; v++) {
        float x = acc[j][v] + bias;
        x = x > 0.0f ? x : 0.0f;               // ReLU
        Hs[mt * 16 + v + 8 * half][nt * 16 + ln] = (_Float16)x;
      }
    }
  }
  __syncthreads();

  // ---- phase 2: E = H @ W2 + b2 -> f16 store ----
  const _Float16* w2r = W2hT + (size_t)r * 256 * 256;
  const float*    b2r = b2 + r * KD;
  {
    const int mt  = wave >> 2;
    const int nt0 = (wave & 3) * 4;
    v8f acc[4] = {{}, {}, {}, {}};
    for (int kb = 0; kb < 256; kb += 32) {
      V16U ua;
      ua.h[0] = *(const v8h*)(&Hs[mt * 16 + ln][kb + 16 * half]);
      ua.h[1] = *(const v8h*)(&Hs[mt * 16 + ln][kb + 16 * half + 8]);
      const _Float16* bbase = w2r + (size_t)(nt0 * 16 + ln) * 256 + kb + 16 * half;
      #pragma unroll
      for (int j = 0; j < 4; j++) {
        v16h bfrag = *(const v16h*)(bbase + (size_t)j * 16 * 256);
        acc[j] = __builtin_amdgcn_wmma_f32_16x16x32_f16(
            false, ua.v, false, bfrag, (short)0, acc[j], false, false);
      }
    }
    #pragma unroll
    for (int j = 0; j < 4; j++) {
      int nt = nt0 + j;
      float bias = b2r[nt * 16 + ln];
      #pragma unroll
      for (int v = 0; v < 8; v++) {
        float x = acc[j][v] + bias;
        int row = rowBase + mt * 16 + v + 8 * half;   // within rule
        outh[((size_t)(r * KB + row)) * KD + nt * 16 + ln] = (_Float16)x;
      }
    }
  }
}

// ---------------- eval + loss reduction ----------------
// one wave (32 lanes) per node; 8 waves per block
__global__ __launch_bounds__(256) void k_eval(
    const _Float16* __restrict__ storeh,
    const float* __restrict__ pos, const float* __restrict__ neg,
    const float* __restrict__ eval_w, const float* __restrict__ eval_b,
    const float* __restrict__ pw, float* __restrict__ sums)
{
  __shared__ float part[5];
  int tid = threadIdx.x;
  if (tid < 5) part[tid] = 0.0f;
  __syncthreads();

  int lane = tid & 31, wave = tid >> 5;
  int node = blockIdx.x * 8 + wave;
  if (node < KTOTAL) {
    uint4 raw = *((const uint4*)(storeh + (size_t)node * KD) + lane);  // 8 halves
    const _Float16* hp = (const _Float16*)&raw;
    float d = 0.0f;
    #pragma unroll
    for (int i = 0; i < 8; i++) d += (float)hp[i] * eval_w[lane * 8 + i];
    #pragma unroll
    for (int off = 16; off > 0; off >>= 1) d += __shfl_xor(d, off, 32);
    if (lane == 0) {
      float x  = d + eval_b[0];
      float pv = pos[node], nv = neg[node];
      float tot = pv + nv;
      if (tot > 0.0f) {
        float tgt = pv / fmaxf(tot, 1e-9f);
        float l1p = log1pf(expf(-fabsf(x)));
        float sp_pos = fmaxf(x, 0.0f) + l1p;     // softplus(x)
        float sp_neg = fmaxf(-x, 0.0f) + l1p;    // softplus(-x)
        float bce = pw[0] * tgt * sp_neg + (1.0f - tgt) * sp_pos;
        atomicAdd(&part[0], tot * bce);
        atomicAdd(&part[1], pv);                 // posTot
        atomicAdd(&part[2], nv);                 // negTot
        if (x >= 0.0f) atomicAdd(&part[3], pv);  // posOK
        else           atomicAdd(&part[4], nv);  // negOK
      }
    }
  }
  __syncthreads();
  if (tid < 5) atomicAdd(&sums[tid], part[tid]);
}

__global__ void k_finalize(const float* __restrict__ sums, float* __restrict__ out) {
  if (threadIdx.x == 0 && blockIdx.x == 0) {
    float loss = sums[0], posTot = sums[1], negTot = sums[2];
    float posOK = sums[3], negOK = sums[4];
    out[0] = loss;
    out[1] = (posTot > 0.0f) ? posOK / fmaxf(posTot, 1e-9f) : 1.0f;
    out[2] = (negTot > 0.0f) ? negOK / fmaxf(negTot, 1e-9f) : 1.0f;
  }
}

// ---------------- host launcher ----------------
extern "C" void kernel_launch(void* const* d_in, const int* in_sizes, int n_in,
                              void* d_out, int out_size, void* d_ws, size_t ws_size,
                              hipStream_t stream) {
  const int*   thax   = (const int*)d_in[0];
  const int*   par    = (const int*)d_in[1];    // [L, NL, 2]
  const float* pos    = (const float*)d_in[2];
  const float* neg    = (const float*)d_in[3];
  const float* table  = (const float*)d_in[4];
  const float* W1     = (const float*)d_in[5];
  const float* b1     = (const float*)d_in[6];
  const float* W2     = (const float*)d_in[7];
  const float* b2     = (const float*)d_in[8];
  const float* evw    = (const float*)d_in[9];
  const float* evb    = (const float*)d_in[10];
  const float* pw     = (const float*)d_in[11];

  char* ws = (char*)d_ws;
  size_t off = 0;
  _Float16* storeh = (_Float16*)(ws + off); off += (size_t)KTOTAL * KD * sizeof(_Float16);
  _Float16* W1hT   = (_Float16*)(ws + off); off += (size_t)KR * 256 * 512 * sizeof(_Float16);
  _Float16* W2hT   = (_Float16*)(ws + off); off += (size_t)KR * 256 * 256 * sizeof(_Float16);
  float*    sums   = (float*)(ws + off);    off += 8 * sizeof(float);

  // prep
  k_conv_w1<<<(KR * 256 * 512 + 255) / 256, 256, 0, stream>>>(W1, W1hT);
  k_conv_w2<<<(KR * 256 * 256 + 255) / 256, 256, 0, stream>>>(W2, W2hT);
  k_init_store<<<(KN0 * KD + 255) / 256, 256, 0, stream>>>(thax, table, storeh);
  k_zero_sums<<<1, 32, 0, stream>>>(sums);

  // layers (topological, sequential on stream)
  for (int l = 0; l < KL; ++l) {
    k_layer<<<dim3(KR, KB / MB), 256, 0, stream>>>(
        par + (size_t)l * KNL * 2,
        storeh,
        storeh + (size_t)(KN0 + l * KNL) * KD,
        W1hT, b1, W2hT, b2);
  }

  // eval + loss
  k_eval<<<KTOTAL / 8, 256, 0, stream>>>(storeh, pos, neg, evw, evb, pw, sums);
  k_finalize<<<1, 32, 0, stream>>>(sums, (float*)d_out);
}
